// Net_33157147525941
// MI455X (gfx1250) — matrix operand order repaired and checked by reading, not compiled, not run on hardware
//
#include <hip/hip_runtime.h>
#include <cstring>

#define NN 131072      // nodes  (32 * 4096)
#define NE 524288      // edges
#define NG 32          // graphs
#define NS 4096        // nodes per graph
#define NW 128         // feature width
static constexpr float FNEG = 1e30f;
static constexpr size_t HS = (size_t)1 << 21;   // hash slots (u64)

typedef __attribute__((ext_vector_type(16))) _Float16 v16h;
typedef __attribute__((ext_vector_type(8)))  float    v8f;

// ------------------------- workspace layout -------------------------
static constexpr size_t SZB      = (size_t)NN * NW * 4;          // 64MB
static constexpr size_t OFF_BIGA = 0;
static constexpr size_t OFF_BIGB = SZB;
static constexpr size_t OFF_BIGC = 2 * SZB;
static constexpr size_t OFF_N2A  = 3 * SZB;                      // [N,2] agg / tx1_0
static constexpr size_t OFF_N2B  = OFF_N2A + (size_t)NN * 2 * 4; // [N,2] gated x
static constexpr size_t OFF_DEG  = OFF_N2B + (size_t)NN * 2 * 4;
static constexpr size_t OFF_SATT = OFF_DEG  + (size_t)NN * 4;
static constexpr size_t OFF_NMA  = OFF_SATT + (size_t)NN * 4;
static constexpr size_t OFF_NMB  = OFF_NMA  + (size_t)NN * 4;
static constexpr size_t OFF_NMC  = OFF_NMB  + (size_t)NN * 4;
static constexpr size_t OFF_REP  = OFF_NMC  + (size_t)NN * 4;
static constexpr size_t OFF_PART = OFF_REP  + (size_t)NN * 4;
static constexpr size_t OFF_PS   = OFF_PART + (size_t)NN * 4;
static constexpr size_t OFF_HSP  = OFF_PS   + (size_t)NN * 4;
static constexpr size_t OFF_AS   = OFF_HSP  + (size_t)NN * 4;
static constexpr size_t OFF_AD   = OFF_AS   + (size_t)NN * 4;
static constexpr size_t OFF_MX   = OFF_AD   + (size_t)NN * 4;
static constexpr size_t OFF_DEN  = OFF_MX   + (size_t)NN * 4;
static constexpr size_t OFF_EMA  = OFF_DEN  + (size_t)NN * 4;
static constexpr size_t OFF_EMB  = OFF_EMA  + (size_t)NE * 4;
static constexpr size_t OFF_EMC  = OFF_EMB  + (size_t)NE * 4;
static constexpr size_t OFF_SRCA = OFF_EMC  + (size_t)NE * 4;
static constexpr size_t OFF_DSTA = OFF_SRCA + (size_t)NE * 4;
static constexpr size_t OFF_SRCB = OFF_DSTA + (size_t)NE * 4;
static constexpr size_t OFF_DSTB = OFF_SRCB + (size_t)NE * 4;
static constexpr size_t OFF_RAW  = OFF_DSTB + (size_t)NE * 4;
static constexpr size_t OFF_SCO  = OFF_RAW  + (size_t)NE * 4;
static constexpr size_t OFF_SKEY = OFF_SCO  + (size_t)NE * 4;
static constexpr size_t OFF_SVAL = OFF_SKEY + (size_t)NE * 4;
static constexpr size_t OFF_HASH = OFF_SVAL + (size_t)NE * 4;   // HS * 8 bytes
static constexpr size_t OFF_STAT = OFF_HASH + HS * 8;           // 257 floats (pad 2048)
static constexpr size_t OFF_GST  = OFF_STAT + 2048;             // mg[32], sume[32]
static constexpr size_t OFF_POOL = OFF_GST  + 256;              // [32,128]
static constexpr size_t OFF_W0HI = OFF_POOL + (size_t)NG * NW * 4;
static constexpr size_t OFF_W0LO = OFF_W0HI + (size_t)NW * NW * 2;
static constexpr size_t OFF_W1HI = OFF_W0LO + (size_t)NW * NW * 2;
static constexpr size_t OFF_W1LO = OFF_W1HI + (size_t)NW * NW * 2;
// total ~248MB of workspace assumed available

// ------------------------- device helpers -------------------------
__device__ __forceinline__ unsigned encSort(float f) {
  unsigned b = __float_as_uint(f);
  return (b & 0x80000000u) ? ~b : (b | 0x80000000u);
}
__device__ __forceinline__ float decSort(unsigned u) {
  unsigned b = (u & 0x80000000u) ? (u & 0x7fffffffu) : ~u;
  return __uint_as_float(b);
}
static unsigned encSortHost(float f) {
  unsigned b; std::memcpy(&b, &f, 4);
  return (b & 0x80000000u) ? ~b : (b | 0x80000000u);
}

// ------------------------- generic fills -------------------------
__global__ void fill_f32(float* p, float v, size_t n) {
  size_t i = (size_t)blockIdx.x * blockDim.x + threadIdx.x;
  size_t st = (size_t)gridDim.x * blockDim.x;
  for (; i < n; i += st) p[i] = v;
}
__global__ void fill_u32(unsigned* p, unsigned v, size_t n) {
  size_t i = (size_t)blockIdx.x * blockDim.x + threadIdx.x;
  size_t st = (size_t)gridDim.x * blockDim.x;
  for (; i < n; i += st) p[i] = v;
}

// ------------------------- GCN + top-k -------------------------
__global__ void k_deg_in(const int* __restrict__ dst, float* __restrict__ deg) {
  int e = blockIdx.x * blockDim.x + threadIdx.x;
  atomicAdd(&deg[dst[e]], 1.0f);
}
__global__ void k_gcn_self(const float* __restrict__ x, const float* __restrict__ deg,
                           float* __restrict__ agg) {
  int i = blockIdx.x * blockDim.x + threadIdx.x;
  float d = rsqrtf(deg[i]);            // deg >= 1 always
  float d2 = d * d;
  agg[2 * i]     = d2 * x[2 * i];
  agg[2 * i + 1] = d2 * x[2 * i + 1];
}
__global__ void k_gcn_edge(const float* __restrict__ x, const int* __restrict__ src,
                           const int* __restrict__ dst, const float* __restrict__ deg,
                           float* __restrict__ agg) {
  int e = blockIdx.x * blockDim.x + threadIdx.x;
  int s = src[e], d = dst[e];
  float c = rsqrtf(deg[s]) * rsqrtf(deg[d]);
  atomicAdd(&agg[2 * d],     c * x[2 * s]);
  atomicAdd(&agg[2 * d + 1], c * x[2 * s + 1]);
}
__global__ void k_gcn_attn(const float* __restrict__ agg, const float* __restrict__ gw,
                           const float* __restrict__ gb, float* __restrict__ sattn) {
  int i = blockIdx.x * blockDim.x + threadIdx.x;
  sattn[i] = agg[2 * i] * gw[0] + agg[2 * i + 1] * gw[1] + gb[0];
}
__global__ void k_graph_stats(const float* __restrict__ s, float* __restrict__ mg,
                              float* __restrict__ se) {
  __shared__ float red[256];
  int g = blockIdx.x, t = threadIdx.x;
  const float* sg = s + (size_t)g * NS;
  float m = -FNEG;
  for (int i = t; i < NS; i += 256) m = fmaxf(m, sg[i]);
  red[t] = m; __syncthreads();
  for (int st = 128; st > 0; st >>= 1) { if (t < st) red[t] = fmaxf(red[t], red[t + st]); __syncthreads(); }
  float mgv = red[0];
  __syncthreads();
  float sum = 0.f;
  for (int i = t; i < NS; i += 256) sum += expf(sg[i] - mgv);
  red[t] = sum; __syncthreads();
  for (int st = 128; st > 0; st >>= 1) { if (t < st) red[t] += red[t + st]; __syncthreads(); }
  if (t == 0) { mg[g] = mgv; se[g] = red[0]; }
}
__global__ void k_topk(const float* __restrict__ x, const float* __restrict__ s,
                       const float* __restrict__ mg, const float* __restrict__ se,
                       float* __restrict__ h0, int* __restrict__ nmask) {
  int i = blockIdx.x * blockDim.x + threadIdx.x;
  int g = i >> 12;
  float sum = se[g];
  float sc = expf(s[i] - mg[g]) / sum;
  float thr = fminf(0.1f, 1.0f / sum);       // mx = exp(0)/sum exactly
  int keep = sc >= thr;
  nmask[i] = keep;
  float w = keep ? sc : 0.f;
  h0[2 * i]     = x[2 * i] * w;
  h0[2 * i + 1] = x[2 * i + 1] * w;
}
__global__ void k_emask1(const int* __restrict__ src, const int* __restrict__ dst,
                         const int* __restrict__ nmask, int* __restrict__ em) {
  int e = blockIdx.x * blockDim.x + threadIdx.x;
  em[e] = nmask[src[e]] && nmask[dst[e]];
}

// ------------------------- Cheb pieces -------------------------
__global__ void k_deg_src(const int* __restrict__ src, const int* __restrict__ em,
                          float* __restrict__ deg) {
  int e = blockIdx.x * blockDim.x + threadIdx.x;
  if (em[e]) atomicAdd(&deg[src[e]], 1.0f);
}
__global__ void k_tx1_small(const float* __restrict__ h0, const int* __restrict__ src,
                            const int* __restrict__ dst, const int* __restrict__ em,
                            const float* __restrict__ deg, float* __restrict__ t0) {
  int e = blockIdx.x * blockDim.x + threadIdx.x;
  if (!em[e]) return;
  int s = src[e], d = dst[e];
  float a = deg[s], b = deg[d];
  float nrm = -((a > 0.f ? rsqrtf(a) : 0.f) * (b > 0.f ? rsqrtf(b) : 0.f));
  if (nrm == 0.f) return;
  atomicAdd(&t0[2 * d],     nrm * h0[2 * s]);
  atomicAdd(&t0[2 * d + 1], nrm * h0[2 * s + 1]);
}
__global__ void k_cheb1(const float* __restrict__ h0, const float* __restrict__ t0,
                        const float* __restrict__ w0, const float* __restrict__ w1,
                        const float* __restrict__ bias, const int* __restrict__ nmask,
                        float* __restrict__ out) {
  size_t idx = (size_t)blockIdx.x * blockDim.x + threadIdx.x;
  int i = (int)(idx >> 7), c = (int)(idx & 127);
  float o = h0[2 * i] * w0[c] + h0[2 * i + 1] * w0[NW + c]
          + t0[2 * i] * w1[c] + t0[2 * i + 1] * w1[NW + c] + bias[c];
  out[idx] = nmask[i] ? fmaxf(o, 0.f) : 0.f;
}
__global__ void k_tx1_big(const float* __restrict__ X, const int* __restrict__ src,
                          const int* __restrict__ dst, const int* __restrict__ em,
                          const float* __restrict__ deg, float* __restrict__ T) {
  int e = blockIdx.x * 8 + (threadIdx.x >> 5);
  int lane = threadIdx.x & 31;
  if (!em[e]) return;
  int s = src[e], d = dst[e];
  float a = deg[s], b = deg[d];
  float nrm = -((a > 0.f ? rsqrtf(a) : 0.f) * (b > 0.f ? rsqrtf(b) : 0.f));
  if (nrm == 0.f) return;
  const float4 x = *(const float4*)(X + (size_t)s * NW + lane * 4);
  float* tp = T + (size_t)d * NW + lane * 4;
  atomicAdd(tp + 0, nrm * x.x);
  atomicAdd(tp + 1, nrm * x.y);
  atomicAdd(tp + 2, nrm * x.z);
  atomicAdd(tp + 3, nrm * x.w);
}
// transpose + hi/lo f16 split of a [128,128] fp32 weight (k-major -> n-major)
__global__ void k_wt_split(const float* __restrict__ w, _Float16* __restrict__ whi,
                           _Float16* __restrict__ wlo) {
  int idx = blockIdx.x * blockDim.x + threadIdx.x;   // 16384
  int k = idx >> 7, n = idx & 127;
  float v = w[k * NW + n];
  _Float16 h = (_Float16)v;
  whi[n * NW + k] = h;
  wlo[n * NW + k] = (_Float16)(v - (float)h);
}

// ------------------------- WMMA GEMM (f16 hi/lo split, fp32 acc) -------------------------
// A-fragment load + hi/lo f16 split for V_WMMA_F32_16X16X32_F16.
// Contiguous k order used here (B matches): lane<16 -> K = ks..ks+15 with ks=k0;
// lane>=16 -> ks = k0+16.
__device__ __forceinline__ void loadA(const float* __restrict__ row, int ks,
                                      v16h& ah, v16h& al) {
  float4 p0 = *(const float4*)(row + ks);
  float4 p1 = *(const float4*)(row + ks + 4);
  float4 p2 = *(const float4*)(row + ks + 8);
  float4 p3 = *(const float4*)(row + ks + 12);
  float xs[16] = { p0.x, p0.y, p0.z, p0.w, p1.x, p1.y, p1.z, p1.w,
                   p2.x, p2.y, p2.z, p2.w, p3.x, p3.y, p3.z, p3.w };
#pragma unroll
  for (int t = 0; t < 16; t++) {
    _Float16 hh = (_Float16)xs[t];
    ah[t] = hh;
    al[t] = (_Float16)(xs[t] - (float)hh);
  }
}

// One wave owns a 16x128 output strip: A converted once per K-step, swept
// across all 8 B column-tiles (8 v8f accumulators = 64 VGPRs).
__global__ void __launch_bounds__(256)
k_cheb_wmma(const float* __restrict__ X, const float* __restrict__ T,
            const _Float16* __restrict__ w0hi, const _Float16* __restrict__ w0lo,
            const _Float16* __restrict__ w1hi, const _Float16* __restrict__ w1lo,
            const float* __restrict__ bias, const int* __restrict__ nmask,
            float* __restrict__ OUT) {
  const int tileM = blockIdx.x * 8 + (threadIdx.x >> 5);   // 0..8191
  const int lane  = threadIdx.x & 31;
  const int half  = lane >> 4;
  const int l15   = lane & 15;
  const int m = tileM * 16 + l15;
  const float* xrow = X + (size_t)m * NW;
  const float* trow = T + (size_t)m * NW;
  __builtin_prefetch(trow, 0, 3);               // global_prefetch_b8
  v8f acc[8];
#pragma unroll
  for (int t = 0; t < 8; t++) acc[t] = (v8f){};

  // pass 1: X @ W0
#pragma unroll
  for (int k0 = 0; k0 < NW; k0 += 32) {
    v16h ah{}, al{};
    loadA(xrow, k0 + half * 16, ah, al);
    const int kb = k0 + half * 16;              // B: 16 consecutive K of one column
    const _Float16* bh0 = w0hi + (size_t)l15 * NW + kb;
    const _Float16* bl0 = w0lo + (size_t)l15 * NW + kb;
#pragma unroll
    for (int tn = 0; tn < 8; tn++) {
      v16h bh = *(const v16h*)(bh0 + (size_t)tn * 16 * NW);
      v16h bl = *(const v16h*)(bl0 + (size_t)tn * 16 * NW);
      acc[tn] = __builtin_amdgcn_wmma_f32_16x16x32_f16(false, ah, false, bh, (short)0, acc[tn], false, false);
      acc[tn] = __builtin_amdgcn_wmma_f32_16x16x32_f16(false, al, false, bh, (short)0, acc[tn], false, false);
      acc[tn] = __builtin_amdgcn_wmma_f32_16x16x32_f16(false, ah, false, bl, (short)0, acc[tn], false, false);
    }
  }
  // pass 2: T @ W1
#pragma unroll
  for (int k0 = 0; k0 < NW; k0 += 32) {
    v16h ah{}, al{};
    loadA(trow, k0 + half * 16, ah, al);
    const int kb = k0 + half * 16;
    const _Float16* bh0 = w1hi + (size_t)l15 * NW + kb;
    const _Float16* bl0 = w1lo + (size_t)l15 * NW + kb;
#pragma unroll
    for (int tn = 0; tn < 8; tn++) {
      v16h bh = *(const v16h*)(bh0 + (size_t)tn * 16 * NW);
      v16h bl = *(const v16h*)(bl0 + (size_t)tn * 16 * NW);
      acc[tn] = __builtin_amdgcn_wmma_f32_16x16x32_f16(false, ah, false, bh, (short)0, acc[tn], false, false);
      acc[tn] = __builtin_amdgcn_wmma_f32_16x16x32_f16(false, al, false, bh, (short)0, acc[tn], false, false);
      acc[tn] = __builtin_amdgcn_wmma_f32_16x16x32_f16(false, ah, false, bl, (short)0, acc[tn], false, false);
    }
  }

  // epilogue: bias + mask + relu.  C layout: VGPR r -> rows r (lanes 0-15) / r+8.
  int nm[8];
#pragma unroll
  for (int r = 0; r < 8; r++) nm[r] = nmask[tileM * 16 + r + half * 8];
#pragma unroll
  for (int tn = 0; tn < 8; tn++) {
    const int n = tn * 16 + l15;
    const float bcol = bias[n];
#pragma unroll
    for (int r = 0; r < 8; r++) {
      int row = tileM * 16 + r + half * 8;
      float v = acc[tn][r] + bcol;
      OUT[(size_t)row * NW + n] = nm[r] ? fmaxf(v, 0.f) : 0.f;
    }
  }
}

// ------------------------- masked batch norm -------------------------
__global__ void k_bn_reduce(const float* __restrict__ X, const int* __restrict__ nmask,
                            float* __restrict__ stats) {   // [0:128]=sum [128:256]=sumsq [256]=cnt
  int c = threadIdx.x;                  // 128 threads
  int base = blockIdx.x * 256;
  float s = 0.f, q = 0.f; int cnt = 0;
  for (int rr = 0; rr < 256; rr++) {
    int row = base + rr;
    int mflag = nmask[row];
    float v = mflag ? X[(size_t)row * NW + c] : 0.f;
    s += v; q += v * v;
    if (c == 0) cnt += mflag;
  }
  atomicAdd(&stats[c], s);
  atomicAdd(&stats[NW + c], q);
  if (c == 0) atomicAdd(&stats[256], (float)cnt);
}
__global__ void k_bn_apply(float* __restrict__ X, const int* __restrict__ nmask,
                           const float* __restrict__ stats, const float* __restrict__ g,
                           const float* __restrict__ b) {
  size_t idx = (size_t)blockIdx.x * blockDim.x + threadIdx.x;
  int i = (int)(idx >> 7), c = (int)(idx & 127);
  float n = fmaxf(stats[256], 1.f);
  float mu = stats[c] / n;
  float var = fmaxf(stats[NW + c] / n - mu * mu, 0.f);
  float m = nmask[i] ? 1.f : 0.f;
  X[idx] = m * ((X[idx] - mu) * rsqrtf(var + 1e-5f) * g[c] + b[c]);
}

// ------------------------- edge pool -------------------------
__global__ void k_node_dots(const float* __restrict__ X, const float* __restrict__ wsv,
                            const float* __restrict__ wdv, float* __restrict__ as_,
                            float* __restrict__ ad_) {
  int i = blockIdx.x * blockDim.x + threadIdx.x;
  const float4* xp = (const float4*)(X + (size_t)i * NW);
  const float4* sp = (const float4*)wsv;
  const float4* dp = (const float4*)wdv;
  float a = 0.f, b = 0.f;
  for (int t = 0; t < NW / 4; t++) {
    float4 x = xp[t], w1 = sp[t], w2 = dp[t];
    a += x.x * w1.x + x.y * w1.y + x.z * w1.z + x.w * w1.w;
    b += x.x * w2.x + x.y * w2.y + x.z * w2.z + x.w * w2.w;
  }
  as_[i] = a; ad_[i] = b;
}
__global__ void k_edge_raw(const float* __restrict__ as_, const float* __restrict__ ad_,
                           const float* __restrict__ bp, const int* __restrict__ src,
                           const int* __restrict__ dst, float* __restrict__ raw) {
  int e = blockIdx.x * blockDim.x + threadIdx.x;
  raw[e] = as_[src[e]] + ad_[dst[e]] + bp[0];
}
__global__ void k_edge_max(const float* __restrict__ raw, const int* __restrict__ em,
                           const int* __restrict__ dst, unsigned* __restrict__ mxu) {
  int e = blockIdx.x * blockDim.x + threadIdx.x;
  if (em[e]) atomicMax(&mxu[dst[e]], encSort(raw[e]));
}
__global__ void k_edge_den(const float* __restrict__ raw, const int* __restrict__ em,
                           const int* __restrict__ dst, const unsigned* __restrict__ mxu,
                           float* __restrict__ den) {
  int e = blockIdx.x * blockDim.x + threadIdx.x;
  if (!em[e]) return;
  float m = decSort(mxu[dst[e]]);
  float mxd = (m > -FNEG * 0.5f) ? m : 0.f;
  atomicAdd(&den[dst[e]], expf(raw[e] - mxd));
}
__global__ void k_edge_score(const float* __restrict__ raw, const int* __restrict__ em,
                             const int* __restrict__ dst, const unsigned* __restrict__ mxu,
                             const float* __restrict__ den, float* __restrict__ score,
                             unsigned* __restrict__ skey, int* __restrict__ sval) {
  int e = blockIdx.x * blockDim.x + threadIdx.x;
  float sc = 0.f;
  if (em[e]) {
    float m = decSort(mxu[dst[e]]);
    float mxd = (m > -FNEG * 0.5f) ? m : 0.f;
    float ee = expf(raw[e] - mxd);
    float dd = den[dst[e]];
    sc = ee / (dd > 0.f ? dd : 1.f) + 0.5f;
  }
  score[e] = sc;
  skey[e] = encSort(em[e] ? -sc : FNEG);
  sval[e] = e;
}
__global__ void k_bitonic(unsigned* __restrict__ key, int* __restrict__ val, int j, int k) {
  int i = blockIdx.x * blockDim.x + threadIdx.x;
  int ixj = i ^ j;
  if (ixj > i) {
    bool up = ((i & k) == 0);
    unsigned a = key[i], b = key[ixj];
    if ((a > b) == up) {
      key[i] = b; key[ixj] = a;
      int t = val[i]; val[i] = val[ixj]; val[ixj] = t;
    }
  }
}
__global__ void k_node_init(int* __restrict__ rep, int* __restrict__ partner,
                            float* __restrict__ ps, int* __restrict__ hsp) {
  int i = blockIdx.x * blockDim.x + threadIdx.x;
  rep[i] = i; partner[i] = -1; ps[i] = 0.f; hsp[i] = 0;
}
// serial greedy matching in score order; merged bitmask lives in LDS (16KB)
__global__ void k_match(const int* __restrict__ order, const int* __restrict__ src,
                        const int* __restrict__ dst, const int* __restrict__ em,
                        const float* __restrict__ score, const int* __restrict__ nmask,
                        int* __restrict__ rep, int* __restrict__ partner,
                        float* __restrict__ ps, int* __restrict__ hsp) {
  __shared__ unsigned mb[NN / 32];
  for (int w = threadIdx.x; w < NN / 32; w += blockDim.x) {
    unsigned bits = 0;
    for (int b = 0; b < 32; b++)
      if (!nmask[w * 32 + b]) bits |= 1u << b;   // not-in-mask = pre-merged
    mb[w] = bits;
  }
  __syncthreads();
  if (threadIdx.x == 0) {
    for (int t = 0; t < NE; t++) {
      int e = order[t];
      if (!em[e]) continue;
      int i = src[e], j = dst[e];
      if (i == j) continue;
      if ((mb[i >> 5] >> (i & 31)) & 1) continue;
      if ((mb[j >> 5] >> (j & 31)) & 1) continue;
      mb[i >> 5] |= 1u << (i & 31);
      mb[j >> 5] |= 1u << (j & 31);
      rep[j] = i; partner[i] = j; ps[i] = score[e]; hsp[i] = 1;
    }
  }
}
__global__ void k_nmask_next(const int* __restrict__ nmI, const int* __restrict__ rep,
                             int* __restrict__ nmO) {
  int i = blockIdx.x * blockDim.x + threadIdx.x;
  nmO[i] = nmI[i] && (rep[i] == i);
}
__global__ void k_merge(const float* __restrict__ Xin, float* __restrict__ Xout,
                        const int* __restrict__ nmI, const int* __restrict__ rep,
                        const int* __restrict__ partner, const float* __restrict__ ps,
                        const int* __restrict__ hsp) {
  size_t idx = (size_t)blockIdx.x * blockDim.x + threadIdx.x;
  int i = (int)(idx >> 7), c = (int)(idx & 127);
  float out = 0.f;
  if (nmI[i] && rep[i] == i) {
    float v = Xin[idx];
    int p = partner[i];
    if (p >= 0) v += Xin[(size_t)p * NW + c];
    out = v * (hsp[i] ? ps[i] : 1.f);
  }
  Xout[idx] = out;
}
__global__ void k_new_edges(const int* __restrict__ srcI, const int* __restrict__ dstI,
                            const int* __restrict__ emI, const int* __restrict__ rep,
                            unsigned long long* __restrict__ tab,
                            int* __restrict__ srcO, int* __restrict__ dstO,
                            int* __restrict__ emO) {
  int e = blockIdx.x * blockDim.x + threadIdx.x;
  int ns = rep[srcI[e]], nd = rep[dstI[e]];
  srcO[e] = ns; dstO[e] = nd;
  int out = 0;
  if (emI[e] && ns != nd) {
    unsigned long long key = (((unsigned long long)ns) << 17) | (unsigned)nd;
    unsigned long long kv = key + 1;
    size_t slot = (size_t)((key * 0x9E3779B97F4A7C15ull) >> 43);   // top 21 bits
    for (;;) {
      unsigned long long old = atomicCAS(&tab[slot], 0ull, kv);
      if (old == 0ull) { out = 1; break; }      // first occurrence wins
      if (old == kv)   { out = 0; break; }      // duplicate pair
      slot = (slot + 1) & (HS - 1);
    }
  }
  emO[e] = out;
}

// ------------------------- head -------------------------
__global__ void k_pool(const float* __restrict__ X, const int* __restrict__ nmask,
                       float* __restrict__ pooled) {
  int g = blockIdx.x, c = threadIdx.x;   // 32 x 128
  float m = -FNEG;
  for (int r = 0; r < NS; r++) {
    int row = g * NS + r;
    if (nmask[row]) m = fmaxf(m, X[(size_t)row * NW + c]);
  }
  pooled[g * NW + c] = m;
}
__global__ void k_head(const float* __restrict__ pooled, const float* __restrict__ lw,
                       const float* __restrict__ lb, float* __restrict__ out) {
  int g = threadIdx.x;
  if (g >= NG) return;
  float z0 = lb[0], z1 = lb[1], z2 = lb[2];
  for (int c = 0; c < NW; c++) {
    float p = pooled[g * NW + c];
    z0 += p * lw[c * 3 + 0]; z1 += p * lw[c * 3 + 1]; z2 += p * lw[c * 3 + 2];
  }
  float mx = fmaxf(z0, fmaxf(z1, z2));
  float lse = logf(expf(z0 - mx) + expf(z1 - mx) + expf(z2 - mx)) + mx;
  out[g * 3 + 0] = z0 - lse; out[g * 3 + 1] = z1 - lse; out[g * 3 + 2] = z2 - lse;
}

// ------------------------- host orchestration -------------------------
static void run_bn(hipStream_t st, float* X, const int* nm, const float* g,
                   const float* b, float* stats) {
  fill_f32<<<4, 256, 0, st>>>(stats, 0.f, 257);
  k_bn_reduce<<<NN / 256, 128, 0, st>>>(X, nm, stats);
  k_bn_apply<<<(NN * NW) / 256, 256, 0, st>>>(X, nm, stats, g, b);
}

static void run_chebbig(hipStream_t st, const float* X, float* T,
                        const int* src, const int* dst, const int* em, const int* nm,
                        const float* w0, const float* w1, const float* bias, float* OUT,
                        float* deg, _Float16* w0hi, _Float16* w0lo,
                        _Float16* w1hi, _Float16* w1lo) {
  fill_f32<<<512, 256, 0, st>>>(deg, 0.f, (size_t)NN);
  k_deg_src<<<NE / 256, 256, 0, st>>>(src, em, deg);
  fill_f32<<<4096, 256, 0, st>>>(T, 0.f, (size_t)NN * NW);
  k_tx1_big<<<NE / 8, 256, 0, st>>>(X, src, dst, em, deg, T);
  k_wt_split<<<64, 256, 0, st>>>(w0, w0hi, w0lo);
  k_wt_split<<<64, 256, 0, st>>>(w1, w1hi, w1lo);
  // 8192 row-tiles, 8 waves (8 row-tiles) per block, each wave does a 16x128 strip
  k_cheb_wmma<<<NN / 16 / 8, 256, 0, st>>>(X, T, w0hi, w0lo, w1hi, w1lo, bias, nm, OUT);
}

static void run_edgepool(hipStream_t st, const float* Xin, float* Xout,
                         const int* srcI, const int* dstI, const int* emI, const int* nmI,
                         int* srcO, int* dstO, int* emO, int* nmO,
                         const float* wsv, const float* wdv, const float* bscal,
                         float* as_, float* ad_, float* raw, float* score,
                         unsigned* mxu, float* den, unsigned* skey, int* sval,
                         int* rep, int* partner, float* ps, int* hsp,
                         unsigned long long* hash) {
  k_node_dots<<<NN / 256, 256, 0, st>>>(Xin, wsv, wdv, as_, ad_);
  k_edge_raw<<<NE / 256, 256, 0, st>>>(as_, ad_, bscal, srcI, dstI, raw);
  fill_u32<<<512, 256, 0, st>>>(mxu, encSortHost(-FNEG), (size_t)NN);
  fill_f32<<<512, 256, 0, st>>>(den, 0.f, (size_t)NN);
  k_edge_max<<<NE / 256, 256, 0, st>>>(raw, emI, dstI, mxu);
  k_edge_den<<<NE / 256, 256, 0, st>>>(raw, emI, dstI, mxu, den);
  k_edge_score<<<NE / 256, 256, 0, st>>>(raw, emI, dstI, mxu, den, score, skey, sval);
  for (int k = 2; k <= NE; k <<= 1)
    for (int j = k >> 1; j > 0; j >>= 1)
      k_bitonic<<<NE / 256, 256, 0, st>>>(skey, sval, j, k);
  k_node_init<<<NN / 256, 256, 0, st>>>(rep, partner, ps, hsp);
  k_match<<<1, 256, 0, st>>>(sval, srcI, dstI, emI, score, nmI, rep, partner, ps, hsp);
  k_nmask_next<<<NN / 256, 256, 0, st>>>(nmI, rep, nmO);
  k_merge<<<(NN * NW) / 256, 256, 0, st>>>(Xin, Xout, nmI, rep, partner, ps, hsp);
  fill_u32<<<2048, 256, 0, st>>>((unsigned*)hash, 0u, HS * 2);
  k_new_edges<<<NE / 256, 256, 0, st>>>(srcI, dstI, emI, rep, hash, srcO, dstO, emO);
}

extern "C" void kernel_launch(void* const* d_in, const int* in_sizes, int n_in,
                              void* d_out, int out_size, void* d_ws, size_t ws_size,
                              hipStream_t stream) {
  (void)in_sizes; (void)n_in; (void)out_size; (void)ws_size;
  const float* x     = (const float*)d_in[0];
  const int*   src0  = (const int*)d_in[1];
  const int*   dst0  = (const int*)d_in[2];
  /* d_in[3] batch_index: derivable as i>>12 */
  const float* gcn_w = (const float*)d_in[4];
  const float* gcn_b = (const float*)d_in[5];
  const float* c1_w0 = (const float*)d_in[6];
  const float* c1_w1 = (const float*)d_in[7];
  const float* c1_b  = (const float*)d_in[8];
  const float* bn1_g = (const float*)d_in[9];
  const float* bn1_b = (const float*)d_in[10];
  const float* c2_w0 = (const float*)d_in[11];
  const float* c2_w1 = (const float*)d_in[12];
  const float* c2_b  = (const float*)d_in[13];
  const float* bn2_g = (const float*)d_in[14];
  const float* bn2_b = (const float*)d_in[15];
  const float* c3_w0 = (const float*)d_in[16];
  const float* c3_w1 = (const float*)d_in[17];
  const float* c3_b  = (const float*)d_in[18];
  const float* bn3_g = (const float*)d_in[19];
  const float* bn3_b = (const float*)d_in[20];
  const float* ep1_ws = (const float*)d_in[21];
  const float* ep1_wd = (const float*)d_in[22];
  const float* ep1_b  = (const float*)d_in[23];
  const float* ep2_ws = (const float*)d_in[24];
  const float* ep2_wd = (const float*)d_in[25];
  const float* ep2_b  = (const float*)d_in[26];
  const float* lin_w  = (const float*)d_in[27];
  const float* lin_b  = (const float*)d_in[28];

  char* ws = (char*)d_ws;
  float* BA   = (float*)(ws + OFF_BIGA);
  float* BB   = (float*)(ws + OFF_BIGB);
  float* BC   = (float*)(ws + OFF_BIGC);
  float* AGG  = (float*)(ws + OFF_N2A);      // also reused as tx1_0
  float* H0   = (float*)(ws + OFF_N2B);
  float* DEG  = (float*)(ws + OFF_DEG);
  float* SATT = (float*)(ws + OFF_SATT);
  int*   NMA  = (int*)(ws + OFF_NMA);
  int*   NMB  = (int*)(ws + OFF_NMB);
  int*   NMC  = (int*)(ws + OFF_NMC);
  int*   REP  = (int*)(ws + OFF_REP);
  int*   PART = (int*)(ws + OFF_PART);
  float* PS   = (float*)(ws + OFF_PS);
  int*   HSP  = (int*)(ws + OFF_HSP);
  float* AS_  = (float*)(ws + OFF_AS);
  float* AD_  = (float*)(ws + OFF_AD);
  unsigned* MXU = (unsigned*)(ws + OFF_MX);
  float* DEN  = (float*)(ws + OFF_DEN);
  int*   EMA  = (int*)(ws + OFF_EMA);
  int*   EMB  = (int*)(ws + OFF_EMB);
  int*   EMC  = (int*)(ws + OFF_EMC);
  int*   SRCA = (int*)(ws + OFF_SRCA);
  int*   DSTA = (int*)(ws + OFF_DSTA);
  int*   SRCB = (int*)(ws + OFF_SRCB);
  int*   DSTB = (int*)(ws + OFF_DSTB);
  float* RAW  = (float*)(ws + OFF_RAW);
  float* SCO  = (float*)(ws + OFF_SCO);
  unsigned* SKEY = (unsigned*)(ws + OFF_SKEY);
  int*   SVAL = (int*)(ws + OFF_SVAL);
  unsigned long long* HASH = (unsigned long long*)(ws + OFF_HASH);
  float* STAT = (float*)(ws + OFF_STAT);
  float* MG   = (float*)(ws + OFF_GST);
  float* SE   = MG + NG;
  float* POOL = (float*)(ws + OFF_POOL);
  _Float16* W0HI = (_Float16*)(ws + OFF_W0HI);
  _Float16* W0LO = (_Float16*)(ws + OFF_W0LO);
  _Float16* W1HI = (_Float16*)(ws + OFF_W1HI);
  _Float16* W1LO = (_Float16*)(ws + OFF_W1LO);

  // ---- GCN attention + top-k gating ----
  fill_f32<<<512, 256, 0, stream>>>(DEG, 1.0f, (size_t)NN);        // self term
  k_deg_in<<<NE / 256, 256, 0, stream>>>(dst0, DEG);
  k_gcn_self<<<NN / 256, 256, 0, stream>>>(x, DEG, AGG);
  k_gcn_edge<<<NE / 256, 256, 0, stream>>>(x, src0, dst0, DEG, AGG);
  k_gcn_attn<<<NN / 256, 256, 0, stream>>>(AGG, gcn_w, gcn_b, SATT);
  k_graph_stats<<<NG, 256, 0, stream>>>(SATT, MG, SE);
  k_topk<<<NN / 256, 256, 0, stream>>>(x, SATT, MG, SE, H0, NMA);
  k_emask1<<<NE / 256, 256, 0, stream>>>(src0, dst0, NMA, EMA);

  // ---- Cheb layer 1 (cin=2, scalar) + BN1 ----
  fill_f32<<<512, 256, 0, stream>>>(DEG, 0.f, (size_t)NN);
  k_deg_src<<<NE / 256, 256, 0, stream>>>(src0, EMA, DEG);
  fill_f32<<<512, 256, 0, stream>>>(AGG, 0.f, (size_t)NN * 2);     // reuse as tx1_0
  k_tx1_small<<<NE / 256, 256, 0, stream>>>(H0, src0, dst0, EMA, DEG, AGG);
  k_cheb1<<<(NN * NW) / 256, 256, 0, stream>>>(H0, AGG, c1_w0, c1_w1, c1_b, NMA, BA);
  run_bn(stream, BA, NMA, bn1_g, bn1_b, STAT);

  // ---- edge pool 1: BA -> BC ----
  run_edgepool(stream, BA, BC, src0, dst0, EMA, NMA, SRCA, DSTA, EMB, NMB,
               ep1_ws, ep1_wd, ep1_b, AS_, AD_, RAW, SCO, MXU, DEN, SKEY, SVAL,
               REP, PART, PS, HSP, HASH);

  // ---- Cheb layer 2 (WMMA) + BN2 ----
  run_chebbig(stream, BC, BB, SRCA, DSTA, EMB, NMB, c2_w0, c2_w1, c2_b, BA,
              DEG, W0HI, W0LO, W1HI, W1LO);
  run_bn(stream, BA, NMB, bn2_g, bn2_b, STAT);

  // ---- edge pool 2: BA -> BC ----
  run_edgepool(stream, BA, BC, SRCA, DSTA, EMB, NMB, SRCB, DSTB, EMC, NMC,
               ep2_ws, ep2_wd, ep2_b, AS_, AD_, RAW, SCO, MXU, DEN, SKEY, SVAL,
               REP, PART, PS, HSP, HASH);

  // ---- Cheb layer 3 (WMMA) + BN3 ----
  run_chebbig(stream, BC, BB, SRCB, DSTB, EMC, NMC, c3_w0, c3_w1, c3_b, BA,
              DEG, W0HI, W0LO, W1HI, W1LO);
  run_bn(stream, BA, NMC, bn3_g, bn3_b, STAT);

  // ---- global max pool + classifier + log_softmax ----
  k_pool<<<NG, NW, 0, stream>>>(BA, NMC, POOL);
  k_head<<<1, 32, 0, stream>>>(POOL, lin_w, lin_b, (float*)d_out);
}
